// SelfAttention_90108413870848
// MI455X (gfx1250) — matrix operand order repaired and checked by reading, not compiled, not run on hardware
//
#include <hip/hip_runtime.h>

// ---------------------------------------------------------------------------
// MI455X (gfx1250, wave32) WMMA self-attention forward.
//   x:[2,2048,1024] f32, Wq/Wk/Wv:[1024,1024], Wo:[1024,1024], bo:[1024]
// ~69 GFLOP vs ~34 MB traffic -> matrix-core bound. f16 WMMA, f32 accumulate.
// proj GEMMs: B tile staged into LDS by the Tensor Data Mover (TENSORcnt).
// out GEMM:   B tile staged via global_load_async_to_lds_b128 (ASYNCcnt).
// Both double-buffered so the DMA overlaps v_wmma issue.
// ---------------------------------------------------------------------------

#define NB     2
#define SEQ    2048
#define DIM    1024
#define HEADS  16
#define HD     64
#define ROWS   (NB * SEQ)          // 4096 global rows

typedef __attribute__((ext_vector_type(16))) _Float16 v16h;
typedef __attribute__((ext_vector_type(8)))  float    v8f;
typedef __attribute__((ext_vector_type(4)))  unsigned tdm_u4;
typedef __attribute__((ext_vector_type(8)))  int      tdm_i8;
typedef __attribute__((ext_vector_type(4)))  int      tdm_i4;

// --- WMMA fragment helpers (contiguous k order, same for A and B) -------
// A 16x32 f16: lane row = lane%16; halves j -> K = hi*16+j
__device__ inline v16h load_a_frag(const _Float16* __restrict__ base_row_k0,
                                   int stride /*halves per row*/, int lo, int hi) {
  const _Float16* p = base_row_k0 + (size_t)lo * stride + hi * 16;
  v16h a;
#pragma unroll
  for (int j = 0; j < 8; ++j) { a[j] = p[j]; a[8 + j] = p[8 + j]; }
  return a;
}

// B 32x16 f16 (column-major source): lane col = lane%16; halves j -> K = hi*16+j
__device__ inline v16h load_b_frag_colmajor(const _Float16* __restrict__ base_col_k0,
                                            size_t stride /*halves per col*/, int lo, int hi) {
  return *(const v16h*)(base_col_k0 + (size_t)lo * stride + hi * 16);
}

__device__ inline v8f wmma_f16(v16h a, v16h b, v8f c) {
  return __builtin_amdgcn_wmma_f32_16x16x32_f16(false, a, false, b, (short)0, c,
                                                false, false);
}

// --- TDM: 2D tile DMA global -> LDS, one wave-level instruction -------------
// Tile: `cols` rows of `kh` contiguous f16 each, row stride `stride_h` halves,
// packed contiguously into LDS at byte offset `lds_byte`.
// D# packing per CDNA5 ISA.
__device__ inline void tdm_load_2d(unsigned lds_byte, const _Float16* gsrc,
                                   int kh, int cols, int stride_h) {
  unsigned long long ga = (unsigned long long)(size_t)gsrc;
  tdm_u4 g0;
  g0[0] = 1u;                                        // count=1, user mode
  g0[1] = lds_byte;                                  // lds_addr [63:32]
  g0[2] = (unsigned)(ga & 0xffffffffu);              // global_addr low
  g0[3] = (unsigned)((ga >> 32) & 0x01ffffffu)       // global_addr[56:32]
          | (2u << 30);                              // type=2 ("image")
  tdm_i8 g1 = {};
  g1[0] = (1 << 16);                                 // data_size=1 -> 2 bytes
  g1[1] = (kh & 0xffff) << 16;                       // tensor_dim0[15:0] @bits63:48
  g1[2] = ((unsigned)kh >> 16) | ((cols & 0xffff) << 16);  // dim0 hi | tensor_dim1 lo
  g1[3] = ((unsigned)cols >> 16) | ((kh & 0xffff) << 16);  // dim1 hi | tile_dim0
  g1[4] = cols & 0xffff;                             // tile_dim1 (tile_dim2=0)
  g1[5] = stride_h;                                  // tensor_dim0_stride low32
  g1[6] = 0;                                         // stride hi | dim1_stride lo
  g1[7] = 0;
  tdm_i4 g2 = {};                                    // 2D: dims 2/3 unused
  tdm_i4 g3 = {};
#if __clang_major__ >= 23
  tdm_i8 g4 = {};
  __builtin_amdgcn_tensor_load_to_lds(g0, g1, g2, g3, g4, 0);
#else
  __builtin_amdgcn_tensor_load_to_lds(g0, g1, g2, g3, 0);
#endif
}
__device__ inline void wait_tensor0() { __builtin_amdgcn_s_wait_tensorcnt(0); }

// --- per-lane async DMA 16B global -> LDS (ASYNCcnt) ------------------------
__device__ inline void async_copy_b128(_Float16* lds_dst, const _Float16* gsrc) {
  unsigned lds_addr = (unsigned)(size_t)lds_dst;
  unsigned long long gaddr = (unsigned long long)(size_t)gsrc;
  asm volatile("global_load_async_to_lds_b128 %0, %1, off"
               :: "v"(lds_addr), "v"(gaddr) : "memory");
}
__device__ inline void wait_async0() {
  asm volatile("s_wait_asynccnt 0x0" ::: "memory");
}

// --- precision conversion ---------------------------------------------------
__global__ void cvt_f32_to_f16(const float* __restrict__ in, _Float16* __restrict__ out,
                               int n) {
  int i = blockIdx.x * blockDim.x + threadIdx.x;
  if (i < n) out[i] = (_Float16)in[i];
}

// out[c*K + k] = in[k*C + c]  (weight transpose so WMMA B-frags are contiguous)
__global__ void cvt_f32_to_f16_T(const float* __restrict__ in, _Float16* __restrict__ out,
                                 int K, int C) {
  int i = blockIdx.x * blockDim.x + threadIdx.x;
  if (i < K * C) {
    int c = i / K, k = i - c * K;
    out[i] = (_Float16)in[(size_t)k * C + c];
  }
}

// ---------------------------------------------------------------------------
// GEMM core: block = 128 threads (4 waves), block tile 64(M) x 64(N), K=1024.
// Wave w owns rows [m0+16w, m0+16w+16). B tile (64 cols x 32 k-halves = 4KB)
// double-buffered in LDS; A frags pipelined in registers.
// ---------------------------------------------------------------------------
#define BS_HALVES (64 * 32)

// out_mode: 0 -> Q/K layout [n,h,s,d] f16; 1 -> V transposed [n,h,d,s] f16
// B staging via Tensor Data Mover (wave 0 issues, TENSORcnt sync).
__global__ __launch_bounds__(128) void proj_gemm(const _Float16* __restrict__ X,
                                                 const _Float16* __restrict__ Wt,
                                                 _Float16* __restrict__ Out,
                                                 int out_mode) {
  __shared__ __align__(16) _Float16 Bs[2][BS_HALVES];
  const int tid = threadIdx.x;
  const int wave = tid >> 5, lane = tid & 31, lo = lane & 15, hi = lane >> 4;
  const int m0 = blockIdx.x * 64 + wave * 16;
  const int n0 = blockIdx.y * 64;

  v8f c[4] = {};
  if (wave == 0)
    tdm_load_2d((unsigned)(size_t)&Bs[0][0], Wt + (size_t)n0 * DIM, 32, 64, DIM);
  v16h a_cur = load_a_frag(X + (size_t)m0 * DIM, DIM, lo, hi);

  for (int k0 = 0; k0 < DIM; k0 += 32) {
    const int buf = (k0 >> 5) & 1;
    if (wave == 0) wait_tensor0();     // stage `buf` fully in LDS
    __syncthreads();                   // visible to all; prior buf^1 reads done
    if (k0 + 32 < DIM && wave == 0)    // DMA next stage, overlaps WMMA below
      tdm_load_2d((unsigned)(size_t)&Bs[buf ^ 1][0],
                  Wt + (size_t)n0 * DIM + k0 + 32, 32, 64, DIM);
    v16h a_next = a_cur;
    if (k0 + 32 < DIM) a_next = load_a_frag(X + (size_t)m0 * DIM + k0 + 32, DIM, lo, hi);
    v16h bfr[4];
#pragma unroll
    for (int t = 0; t < 4; ++t)        // issue all LDS reads, then all WMMAs
      bfr[t] = *(const v16h*)(&Bs[buf][(t * 16 + lo) * 32 + hi * 16]);
#pragma unroll
    for (int t = 0; t < 4; ++t) c[t] = wmma_f16(a_cur, bfr[t], c[t]);
    a_cur = a_next;
  }

#pragma unroll
  for (int t = 0; t < 4; ++t) {
#pragma unroll
    for (int r = 0; r < 8; ++r) {
      int row = m0 + r + 8 * hi;             // global row in [0,4096)
      int col = n0 + t * 16 + lo;            // inner dim in [0,1024)
      int n = row >> 11, s = row & (SEQ - 1);
      int h = col >> 6,  d = col & (HD - 1);
      size_t idx = out_mode == 0
                       ? (((size_t)(n * HEADS + h) * SEQ + s) * HD + d)      // [n,h,s,d]
                       : (((size_t)(n * HEADS + h) * HD + d) * SEQ + s);     // [n,h,d,s]
      Out[idx] = (_Float16)c[t][r];
    }
  }
}

// --- output projection: out = A[4096x1024] * Wo + bo  (f32 result) ----------
// B staging via per-lane async loads (ASYNCcnt path kept alongside TDM).
__device__ inline void stage_B_async(int tid, _Float16* buf,
                                     const _Float16* __restrict__ Wt, int n0, int k0) {
#pragma unroll
  for (int c2 = 0; c2 < 2; ++c2) {
    int id = tid + c2 * 128;                 // 256 chunks of 8 halves (16 B)
    int c = id >> 2, p = id & 3;
    async_copy_b128(buf + c * 32 + p * 8,
                    Wt + (size_t)(n0 + c) * DIM + k0 + p * 8);
  }
}

__global__ __launch_bounds__(128) void out_gemm(const _Float16* __restrict__ A,
                                                const _Float16* __restrict__ Wot,
                                                const float* __restrict__ bo,
                                                float* __restrict__ Out) {
  __shared__ __align__(16) _Float16 Bs[2][BS_HALVES];
  const int tid = threadIdx.x;
  const int wave = tid >> 5, lane = tid & 31, lo = lane & 15, hi = lane >> 4;
  const int m0 = blockIdx.x * 64 + wave * 16;
  const int n0 = blockIdx.y * 64;

  v8f c[4] = {};
  stage_B_async(tid, Bs[0], Wot, n0, 0);
  v16h a_cur = load_a_frag(A + (size_t)m0 * DIM, DIM, lo, hi);

  for (int k0 = 0; k0 < DIM; k0 += 32) {
    const int buf = (k0 >> 5) & 1;
    wait_async0();
    __syncthreads();
    if (k0 + 32 < DIM) stage_B_async(tid, Bs[buf ^ 1], Wot, n0, k0 + 32);
    v16h a_next = a_cur;
    if (k0 + 32 < DIM) a_next = load_a_frag(A + (size_t)m0 * DIM + k0 + 32, DIM, lo, hi);
    v16h bfr[4];
#pragma unroll
    for (int t = 0; t < 4; ++t)
      bfr[t] = *(const v16h*)(&Bs[buf][(t * 16 + lo) * 32 + hi * 16]);
#pragma unroll
    for (int t = 0; t < 4; ++t) c[t] = wmma_f16(a_cur, bfr[t], c[t]);
    a_cur = a_next;
  }

#pragma unroll
  for (int t = 0; t < 4; ++t)
#pragma unroll
    for (int r = 0; r < 8; ++r) {
      int row = m0 + r + 8 * hi;
      int col = n0 + t * 16 + lo;
      Out[(size_t)row * DIM + col] = c[t][r] + bo[col];
    }
}

// --- flash attention: one wave per (n, h, 16-row q tile) --------------------
__global__ __launch_bounds__(32) void flash_attn(const _Float16* __restrict__ Qh,
                                                 const _Float16* __restrict__ Kh,
                                                 const _Float16* __restrict__ Vt,
                                                 _Float16* __restrict__ Ah) {
  __shared__ __align__(16) _Float16 lds_p[16 * 32];  // P tile (C-layout -> A-layout)
  const int lane = threadIdx.x, lo = lane & 15, hi = lane >> 4;
  const int q0 = blockIdx.x * 16;
  const int h  = blockIdx.y;
  const int n  = blockIdx.z;
  const float scale = 0.125f;                // 1/sqrt(64)

  const _Float16* qb = Qh + (size_t)(n * HEADS + h) * SEQ * HD;
  const _Float16* kb = Kh + (size_t)(n * HEADS + h) * SEQ * HD;
  const _Float16* vb = Vt + (size_t)(n * HEADS + h) * HD * SEQ;

  // Q A-frags (16 rows x 64 d = two K=32 fragments), loaded once
  v16h qa0 = load_a_frag(qb + (size_t)q0 * HD + 0,  HD, lo, hi);
  v16h qa1 = load_a_frag(qb + (size_t)q0 * HD + 32, HD, lo, hi);

  v8f o[4] = {};
  float mprev[8], lprev[8];
#pragma unroll
  for (int r = 0; r < 8; ++r) { mprev[r] = -3.0e38f; lprev[r] = 0.0f; }

  for (int j0 = 0; j0 < q0 + 16; j0 += 32) {   // causal: keys <= q0+15
    // ---- prefetch ALL K and V fragments for this 32-key step up front ----
    v16h kf[4], vf[4];
    kf[0] = load_b_frag_colmajor(kb + (size_t)j0 * HD + 0,         HD, lo, hi);
    kf[1] = load_b_frag_colmajor(kb + (size_t)j0 * HD + 32,        HD, lo, hi);
    kf[2] = load_b_frag_colmajor(kb + (size_t)(j0 + 16) * HD + 0,  HD, lo, hi);
    kf[3] = load_b_frag_colmajor(kb + (size_t)(j0 + 16) * HD + 32, HD, lo, hi);
#pragma unroll
    for (int t = 0; t < 4; ++t)                // V loads hide under QK^T + softmax
      vf[t] = *(const v16h*)(vb + (size_t)(t * 16 + lo) * SEQ + j0 + hi * 16);

    // ---- logits for 32 keys: two 16x16 tiles, K-dim = d (64) ----
    v8f s0 = {}, s1 = {};
    s0 = wmma_f16(qa0, kf[0], s0);
    s0 = wmma_f16(qa1, kf[1], s0);
    s1 = wmma_f16(qa0, kf[2], s1);
    s1 = wmma_f16(qa1, kf[3], s1);

    // ---- scale + causal mask + row max (C layout: M = r+8*hi, N = lo) ----
    float rmax[8];
#pragma unroll
    for (int r = 0; r < 8; ++r) {
      int qrow = q0 + r + 8 * hi;
      float v0 = s0[r] * scale; if (j0 + lo      > qrow) v0 = -3.0e38f;
      float v1 = s1[r] * scale; if (j0 + 16 + lo > qrow) v1 = -3.0e38f;
      s0[r] = v0; s1[r] = v1;
      rmax[r] = fmaxf(v0, v1);
    }
#pragma unroll
    for (int r = 0; r < 8; ++r)
      for (int m = 1; m < 16; m <<= 1)
        rmax[r] = fmaxf(rmax[r], __shfl_xor(rmax[r], m, 32));

    // ---- online softmax update ----
    float alpha[8], p0[8], p1[8], rs[8];
#pragma unroll
    for (int r = 0; r < 8; ++r) {
      float mn = fmaxf(mprev[r], rmax[r]);
      alpha[r] = __expf(mprev[r] - mn);
      mprev[r] = mn;
      p0[r] = __expf(s0[r] - mn);
      p1[r] = __expf(s1[r] - mn);
      rs[r] = p0[r] + p1[r];
    }
#pragma unroll
    for (int r = 0; r < 8; ++r)
      for (int m = 1; m < 16; m <<= 1)
        rs[r] += __shfl_xor(rs[r], m, 32);
#pragma unroll
    for (int r = 0; r < 8; ++r) lprev[r] = lprev[r] * alpha[r] + rs[r];
#pragma unroll
    for (int t = 0; t < 4; ++t)
#pragma unroll
      for (int r = 0; r < 8; ++r) o[t][r] *= alpha[r];

    // ---- stage P (16x32) through LDS to reshape into A-fragment ----
#pragma unroll
    for (int r = 0; r < 8; ++r) {
      int row = r + 8 * hi;
      lds_p[row * 32 + lo]      = (_Float16)p0[r];
      lds_p[row * 32 + 16 + lo] = (_Float16)p1[r];
    }
    __syncthreads();
    v16h pa = load_a_frag(lds_p, 32, lo, hi);

    // ---- O += P(16x32) * V(32x64); V frags already in registers ----
#pragma unroll
    for (int t = 0; t < 4; ++t) o[t] = wmma_f16(pa, vf[t], o[t]);
    __syncthreads();
  }

  // ---- finalize: O /= l, write attn out [n, s, h*64+d] as f16 ----
  float inv[8];
#pragma unroll
  for (int r = 0; r < 8; ++r) inv[r] = 1.0f / lprev[r];
#pragma unroll
  for (int t = 0; t < 4; ++t)
#pragma unroll
    for (int r = 0; r < 8; ++r) {
      int s = q0 + r + 8 * hi;
      int col = h * HD + t * 16 + lo;
      Ah[((size_t)n * SEQ + s) * DIM + col] = (_Float16)(o[t][r] * inv[r]);
    }
}

// ---------------------------------------------------------------------------
extern "C" void kernel_launch(void* const* d_in, const int* in_sizes, int n_in,
                              void* d_out, int out_size, void* d_ws, size_t ws_size,
                              hipStream_t stream) {
  const float* x  = (const float*)d_in[0];
  const float* Wq = (const float*)d_in[1];
  const float* Wk = (const float*)d_in[2];
  const float* Wv = (const float*)d_in[3];
  const float* Wo = (const float*)d_in[4];
  const float* bo = (const float*)d_in[5];
  float* out = (float*)d_out;

  _Float16* ws = (_Float16*)d_ws;
  const size_t NX = (size_t)ROWS * DIM;          // 4M halves
  const size_t NW = (size_t)DIM * DIM;           // 1M halves
  _Float16* xh  = ws;                            // x as f16 [4096,1024]
  _Float16* wqt = xh  + NX;                      // W^T f16 [1024(col),1024(k)]
  _Float16* wkt = wqt + NW;
  _Float16* wvt = wkt + NW;
  _Float16* wot = wvt + NW;
  _Float16* Qh  = wot + NW;                      // [n,h,s,d]
  _Float16* Kh  = Qh  + NX;                      // [n,h,s,d]
  _Float16* Vt  = Kh  + NX;                      // [n,h,d,s]
  _Float16* Ah  = Vt  + NX;                      // attn out [n,s,h*d] f16

  // 1) precision conversion (+ weight transpose for contiguous B-fragments)
  cvt_f32_to_f16<<<(int)((NX + 255) / 256), 256, 0, stream>>>(x, xh, (int)NX);
  cvt_f32_to_f16_T<<<(int)((NW + 255) / 256), 256, 0, stream>>>(Wq, wqt, DIM, DIM);
  cvt_f32_to_f16_T<<<(int)((NW + 255) / 256), 256, 0, stream>>>(Wk, wkt, DIM, DIM);
  cvt_f32_to_f16_T<<<(int)((NW + 255) / 256), 256, 0, stream>>>(Wv, wvt, DIM, DIM);
  cvt_f32_to_f16_T<<<(int)((NW + 255) / 256), 256, 0, stream>>>(Wo, wot, DIM, DIM);

  // 2) Q/K/V projections (WMMA GEMM, TDM-staged B), 64x64 per block
  dim3 pg(ROWS / 64, DIM / 64);
  proj_gemm<<<pg, 128, 0, stream>>>(xh, wqt, Qh, 0);
  proj_gemm<<<pg, 128, 0, stream>>>(xh, wkt, Kh, 0);
  proj_gemm<<<pg, 128, 0, stream>>>(xh, wvt, Vt, 1);   // V transposed layout

  // 3) causal flash attention, one wave per (q-tile, head, batch)
  dim3 ag(SEQ / 16, HEADS, NB);
  flash_attn<<<ag, 32, 0, stream>>>(Qh, Kh, Vt, Ah);

  // 4) output projection + bias, f32 result (ASYNCcnt-staged B)
  out_gemm<<<pg, 128, 0, stream>>>(Ah, wot, bo, out);
}
